// MultiHeadLatentAttention_28054726377699
// MI455X (gfx1250) — matrix operand order repaired and checked by reading, not compiled, not run on hardware
//
#include <hip/hip_runtime.h>
#include <hip/hip_bf16.h>

typedef __attribute__((ext_vector_type(16))) __bf16 v16bf;
typedef __attribute__((ext_vector_type(8)))  __bf16 v8bf;
typedef __attribute__((ext_vector_type(8)))  float  v8f;
typedef unsigned short u16;
typedef unsigned int   u32;

constexpr int kB   = 2;
constexpr int kS   = 2048;
constexpr int kH   = 16;
constexpr int kD   = 128;      // head dim
constexpr int kHID = 2048;
constexpr int kLAT = 512;
constexpr int kM   = kB * kS;  // 4096 rows for all projections

constexpr int KT16 = 4 * 16 * 32;  // halfs per packed 16-key K tile (4 chunks x 16 keys x 32 dims)
constexpr int VT32 = 8 * 16 * 32;  // halfs per packed 32-key V tile (8 dimtiles x 16 dims x 32 keys)

#define ATT_SCALE 0.08838834764831845f  /* 1/sqrt(128) */

__device__ __forceinline__ u16 f2bf(float f) {
  u32 u = __float_as_uint(f);
  u32 r = u + 0x7FFFu + ((u >> 16) & 1u);  // round-to-nearest-even
  return (u16)(r >> 16);
}
__device__ __forceinline__ float bf2f(u16 h) {
  return __uint_as_float(((u32)h) << 16);
}

// Build a 16-half fragment from two contiguous 8-half (16B) chunks.
__device__ __forceinline__ v16bf ldfrag2(const u16* p0, const u16* p1) {
  v8bf a = *(const v8bf*)p0;
  v8bf b = *(const v8bf*)p1;
  v16bf r;
#pragma unroll
  for (int i = 0; i < 8; ++i) { r[i] = a[i]; r[i + 8] = b[i]; }
  return r;
}

// ---------------------------------------------------------------- converts
__global__ void f2bf_kernel(const float* __restrict__ src, u16* __restrict__ dst, int n) {
  int i = blockIdx.x * blockDim.x + threadIdx.x;
  if (i < n) dst[i] = f2bf(src[i]);
}

// src [K][N] fp32 row-major  ->  dst [N][K] bf16 (pre-transposed weights)
__global__ void transpose_f2bf(const float* __restrict__ src, u16* __restrict__ dst,
                               int Kd, int Nd) {
  int i = blockIdx.x * blockDim.x + threadIdx.x;
  if (i >= Kd * Nd) return;
  int k = i / Nd;
  int n = i - k * Nd;
  dst[(size_t)n * Kd + k] = f2bf(src[i]);
}

// ---------------------------------------------------------------- GEMM
// C[M][N] = A[M][K] @ Wt[N][K]^T + bias, bf16 operands, f32 accumulate.
// Block 256 threads = 8 waves; block tile 128x128; wave tile 32x64.
template <bool OUT_BF16>
__global__ __launch_bounds__(256)
void gemm_bf16_wmma(const u16* __restrict__ A, const u16* __restrict__ Wt,
                    const float* __restrict__ bias, void* __restrict__ Cout,
                    int M, int N, int K) {
  const int lane = threadIdx.x & 31;
  const int wave = threadIdx.x >> 5;
  const int lrow = lane & 15;
  const bool hiw = lane >= 16;

  const int mW = blockIdx.x * 128 + (wave & 3) * 32;
  const int nW = blockIdx.y * 128 + (wave >> 2) * 64;

  const v8f vz = {0.f, 0.f, 0.f, 0.f, 0.f, 0.f, 0.f, 0.f};
  v8f acc[2][4];
#pragma unroll
  for (int i = 0; i < 2; ++i)
#pragma unroll
    for (int j = 0; j < 4; ++j) acc[i][j] = vz;

  for (int k0 = 0; k0 < K; k0 += 32) {
    // A fragments: lane<16 -> K {0..7,16..23}; lane>=16 -> K {8..15,24..31}
    v16bf af[2];
#pragma unroll
    for (int i = 0; i < 2; ++i) {
      const u16* p = A + (size_t)(mW + i * 16 + lrow) * K + k0 + (hiw ? 8 : 0);
      af[i] = ldfrag2(p, p + 16);
    }
    // B fragments from Wt[N][K]: lane<16 -> K {0..7,16..23}; lane>=16 -> K {8..15,24..31}
    v16bf bfr[4];
#pragma unroll
    for (int j = 0; j < 4; ++j) {
      const u16* p = Wt + (size_t)(nW + j * 16 + lrow) * K + k0 + (hiw ? 8 : 0);
      bfr[j] = ldfrag2(p, p + 16);
    }
#pragma unroll
    for (int i = 0; i < 2; ++i)
#pragma unroll
      for (int j = 0; j < 4; ++j)
        acc[i][j] = __builtin_amdgcn_wmma_f32_16x16x32_bf16(
            false, af[i], false, bfr[j], (short)0, acc[i][j], false, false);
  }

#pragma unroll
  for (int i = 0; i < 2; ++i)
#pragma unroll
    for (int j = 0; j < 4; ++j) {
      const int col = nW + j * 16 + lrow;
      const float bb = bias[col];
#pragma unroll
      for (int r = 0; r < 8; ++r) {
        const int row = mW + i * 16 + r + (hiw ? 8 : 0);
        const float v = acc[i][j][r] + bb;
        if (OUT_BF16)
          ((u16*)Cout)[(size_t)row * N + col] = f2bf(v);
        else
          ((float*)Cout)[(size_t)row * N + col] = v;
      }
    }
}

// ---------------------------------------------------------------- RoPE + pack
// Builds Q [B][S][H][128] (nope|rope) and Kp in B-fragment-native packed layout:
//   Kp[b][h][keytile16][chunk(4)][key%16][dim%32]   (contiguous 32-dim rows per lane)
__global__ void rope_pack(const u16* __restrict__ q1, const u16* __restrict__ q2,
                          const u16* __restrict__ k1, const u16* __restrict__ k2,
                          u16* __restrict__ Qb, u16* __restrict__ Kp) {
  int idx = blockIdx.x * blockDim.x + threadIdx.x;
  if (idx >= kB * kS * kH * 64) return;
  const int j = idx & 63;            idx >>= 6;
  const int h = idx & (kH - 1);      idx >>= 4;
  const int s = idx & (kS - 1);      idx >>= 11;
  const int b = idx;

  const size_t base = (size_t)(b * kS + s) * (kH * 64) + h * 64;
  const int jp = (j < 32) ? j : j - 32;
  // inv_freq = 10000^(-(2*jp)/64)
  const float inv = __expf(-((float)(2 * jp) * (1.0f / 64.f)) * 9.210340371976184f);
  float sn, cs;
  __sincosf((float)s * inv, &sn, &cs);

  const float q_rot = bf2f(q2[base + jp]);
  const float q_pas = bf2f(q2[base + jp + 32]);
  const float k_rot = bf2f(k2[base + jp]);
  const float k_pas = bf2f(k2[base + jp + 32]);
  const float qr = (j < 32) ? (q_rot * cs - q_pas * sn) : (q_rot * sn + q_pas * cs);
  const float kr = (j < 32) ? (k_rot * cs - k_pas * sn) : (k_rot * sn + k_pas * cs);

  const size_t qrow = ((size_t)(b * kS + s) * kH + h) * kD;
  Qb[qrow + j]      = q1[base + j];   // nope half (already bf16)
  Qb[qrow + 64 + j] = f2bf(qr);       // rope half

  // packed K: tile=s>>4, chunk=d>>5, key-in-tile=s&15, dim-in-chunk=d&31
  const size_t kbh = ((size_t)(b * kH + h) * (kS / 16) + (s >> 4));
  const int d0 = j, d1 = 64 + j;
  Kp[((kbh * 4 + (d0 >> 5)) * 16 + (s & 15)) * 32 + (d0 & 31)] = k1[base + j];
  Kp[((kbh * 4 + (d1 >> 5)) * 16 + (s & 15)) * 32 + (d1 & 31)] = f2bf(kr);
}

// V [B][S][H][128] -> Vp[b][h][keytile32][dimtile(8)][dim%16][key%32]
__global__ void v_pack(const u16* __restrict__ Vsrc, u16* __restrict__ Vp) {
  int idx = blockIdx.x * blockDim.x + threadIdx.x;
  if (idx >= kB * kS * kH * kD) return;
  const u16 v = Vsrc[idx];
  const int d = idx & 127;           int r = idx >> 7;
  const int h = r & (kH - 1);        r >>= 4;
  const int s = r & (kS - 1);        r >>= 11;
  const int b = r;
  const size_t vbh = ((size_t)(b * kH + h) * (kS / 32) + (s >> 5));
  Vp[((vbh * 8 + (d >> 4)) * 16 + (d & 15)) * 32 + (s & 31)] = v;
}

// ---------------------------------------------------------------- flash attention
// One wave per 16-row q-tile; 32-key tiles; online softmax; all matmuls via WMMA.
// K/V operands come from fragment-native packed buffers -> pure b128 loads.
__global__ __launch_bounds__(128)
void mla_attn(const u16* __restrict__ Q,   // [B][S][H][128]
              const u16* __restrict__ Kp,  // packed, see rope_pack
              const u16* __restrict__ Vp,  // packed, see v_pack
              u16* __restrict__ O) {       // [B][S][H][128]
  __shared__ u16 plds[4][16 * 40];  // per-wave P staging, row stride 40 halfs (16B aligned)
  const int lane = threadIdx.x & 31;
  const int wave = threadIdx.x >> 5;
  const int lrow = lane & 15;
  const bool hiw = lane >= 16;

  int tile = blockIdx.x * 4 + wave;
  const int qt = tile & (kS / 16 - 1); tile >>= 7;
  const int h  = tile & (kH - 1);      tile >>= 4;
  const int b  = tile;
  const int qbase = qt * 16;

  // Q fragments across 4 head-dim chunks of 32
  const u16* qrow = Q + ((size_t)(b * kS + qbase + lrow) * kH + h) * kD;
  v16bf qf[4];
#pragma unroll
  for (int c = 0; c < 4; ++c) {
    const u16* p = qrow + c * 32 + (hiw ? 8 : 0);
    qf[c] = ldfrag2(p, p + 16);
  }

  const v8f vz = {0.f, 0.f, 0.f, 0.f, 0.f, 0.f, 0.f, 0.f};
  v8f o[8];
#pragma unroll
  for (int dt = 0; dt < 8; ++dt) o[dt] = vz;
  float m_[8], l_[8];
#pragma unroll
  for (int r = 0; r < 8; ++r) { m_[r] = -__builtin_inff(); l_[r] = 0.f; }

  const u16* kpb = Kp + (size_t)(b * kH + h) * (kS / 16) * KT16;
  const u16* vpb = Vp + (size_t)(b * kH + h) * (kS / 32) * VT32;
  const int ntiles = (qbase + 47) >> 5;  // cover keys 0 .. qbase+15

  for (int t = 0; t < ntiles; ++t) {
    // prefetch next key tile (K: 8KB, V: 8KB) while this one computes
    if (t + 1 < ntiles) {
      const u16* kn = kpb + (size_t)(2 * (t + 1)) * KT16;
      const u16* vn = vpb + (size_t)(t + 1) * VT32;
      __builtin_prefetch(kn + (size_t)lane * 64, 0, 3);
      __builtin_prefetch(kn + 2048 + (size_t)lane * 64, 0, 3);
      __builtin_prefetch(vn + (size_t)lane * 64, 0, 3);
      __builtin_prefetch(vn + 2048 + (size_t)lane * 64, 0, 3);
    }

    const int kt0 = t * 32;
    const u16* kt_a = kpb + (size_t)(2 * t) * KT16;      // keys kt0 .. kt0+15
    const u16* kt_b = kt_a + KT16;                       // keys kt0+16 .. kt0+31
    v8f s0 = vz, s1 = vz;
#pragma unroll
    for (int c = 0; c < 4; ++c) {
      const u16* p0 = kt_a + ((size_t)(c * 16 + lrow)) * 32 + (hiw ? 8 : 0);
      const u16* p1 = kt_b + ((size_t)(c * 16 + lrow)) * 32 + (hiw ? 8 : 0);
      v16bf b0 = ldfrag2(p0, p0 + 16);
      v16bf b1 = ldfrag2(p1, p1 + 16);
      s0 = __builtin_amdgcn_wmma_f32_16x16x32_bf16(false, qf[c], false, b0, (short)0, s0, false, false);
      s1 = __builtin_amdgcn_wmma_f32_16x16x32_bf16(false, qf[c], false, b1, (short)0, s1, false, false);
    }

    // online softmax (row lives across the 16 lanes of each half-wave group)
    float p0v[8], p1v[8], corr[8];
#pragma unroll
    for (int r = 0; r < 8; ++r) {
      const int row = qbase + r + (hiw ? 8 : 0);
      const int kc0 = kt0 + lrow;
      float a0 = (kc0      <= row) ? s0[r] * ATT_SCALE : -__builtin_inff();
      float a1 = (kc0 + 16 <= row) ? s1[r] * ATT_SCALE : -__builtin_inff();
      float rmax = fmaxf(a0, a1);
#pragma unroll
      for (int msk = 1; msk < 16; msk <<= 1)
        rmax = fmaxf(rmax, __shfl_xor(rmax, msk, 32));
      const float mn = fmaxf(m_[r], rmax);
      const float cr = __expf(m_[r] - mn);
      const float e0 = __expf(a0 - mn);
      const float e1 = __expf(a1 - mn);
      float rsum = e0 + e1;
#pragma unroll
      for (int msk = 1; msk < 16; msk <<= 1)
        rsum += __shfl_xor(rsum, msk, 32);
      l_[r] = l_[r] * cr + rsum;
      m_[r] = mn;
      corr[r] = cr;
      p0v[r] = e0;
      p1v[r] = e1;
    }

    // V fragments: issue the b128 loads early, before the LDS fence
    const u16* vt = vpb + (size_t)t * VT32;
    v16bf vf[8];
#pragma unroll
    for (int dt = 0; dt < 8; ++dt) {
      const u16* pv = vt + ((size_t)(dt * 16 + lrow)) * 32 + (hiw ? 8 : 0);
      vf[dt] = ldfrag2(pv, pv + 16);
    }

#pragma unroll
    for (int dt = 0; dt < 8; ++dt)
#pragma unroll
      for (int r = 0; r < 8; ++r) o[dt][r] *= corr[r];

    // P: C-layout -> LDS row-major -> reload in A-layout (wave-private, in-order DS)
    u16* pl = plds[wave];
    const int rbase = hiw ? 8 : 0;
#pragma unroll
    for (int r = 0; r < 8; ++r) {
      pl[(r + rbase) * 40 + lrow]      = f2bf(p0v[r]);
      pl[(r + rbase) * 40 + 16 + lrow] = f2bf(p1v[r]);
    }
    asm volatile("s_wait_dscnt 0" ::: "memory");
    const u16* pr = pl + lrow * 40 + (hiw ? 8 : 0);
    v16bf pa = ldfrag2(pr, pr + 16);
    asm volatile("" ::: "memory");

    // O += P @ V
#pragma unroll
    for (int dt = 0; dt < 8; ++dt)
      o[dt] = __builtin_amdgcn_wmma_f32_16x16x32_bf16(false, pa, false, vf[dt], (short)0, o[dt], false, false);
  }

#pragma unroll
  for (int dt = 0; dt < 8; ++dt)
#pragma unroll
    for (int r = 0; r < 8; ++r) {
      const int row = qbase + r + (hiw ? 8 : 0);
      const float val = o[dt][r] / l_[r];
      O[((size_t)(b * kS + row) * kH + h) * kD + dt * 16 + lrow] = f2bf(val);
    }
}

// ---------------------------------------------------------------- launch
extern "C" void kernel_launch(void* const* d_in, const int* in_sizes, int n_in,
                              void* d_out, int out_size, void* d_ws, size_t ws_size,
                              hipStream_t stream) {
  const float* x      = (const float*)d_in[0];
  // d_in[1] = attention_mask (tril ones) -> implemented analytically as causal mask
  const float* W_kv_d = (const float*)d_in[2];
  const float* b_kv_d = (const float*)d_in[3];
  const float* W_q_d  = (const float*)d_in[4];
  const float* b_q_d  = (const float*)d_in[5];
  const float* W_k_u  = (const float*)d_in[6];
  const float* b_k_u  = (const float*)d_in[7];
  const float* W_q_u  = (const float*)d_in[8];
  const float* b_q_u  = (const float*)d_in[9];
  const float* W_v_u  = (const float*)d_in[10];
  const float* b_v_u  = (const float*)d_in[11];
  const float* W_rk   = (const float*)d_in[12];
  const float* b_rk   = (const float*)d_in[13];
  const float* W_rq   = (const float*)d_in[14];
  const float* b_rq   = (const float*)d_in[15];
  const float* W_o    = (const float*)d_in[16];
  const float* b_o    = (const float*)d_in[17];

  char* ws = (char*)d_ws;
  size_t off = 0;
  auto alloc16 = [&](size_t elems) -> u16* {
    u16* p = (u16*)(ws + off);
    off += (elems * 2 + 255) & ~(size_t)255;
    return p;
  };

  u16* xb     = alloc16((size_t)kM * kHID);
  u16* wt_kvd = alloc16((size_t)kHID * kLAT);   // [512][2048]
  u16* wt_qd  = alloc16((size_t)kHID * kLAT);
  u16* wt_ku  = alloc16((size_t)kLAT * 1024);   // [1024][512]
  u16* wt_qu  = alloc16((size_t)kLAT * 1024);
  u16* wt_vu  = alloc16((size_t)kLAT * 2048);   // [2048][512]
  u16* wt_rk  = alloc16((size_t)kHID * 1024);   // [1024][2048]
  u16* wt_rq  = alloc16((size_t)kLAT * 1024);
  u16* wt_o   = alloc16((size_t)kHID * kHID);   // [2048][2048]
  u16* kvd    = alloc16((size_t)kM * kLAT);
  u16* qd     = alloc16((size_t)kM * kLAT);
  u16* k1b    = alloc16((size_t)kM * 1024);
  u16* q1b    = alloc16((size_t)kM * 1024);
  u16* vb     = alloc16((size_t)kM * 2048);
  u16* k2b    = alloc16((size_t)kM * 1024);
  u16* q2b    = alloc16((size_t)kM * 1024);
  u16* qbuf   = alloc16((size_t)kM * 2048);
  u16* kpbuf  = alloc16((size_t)kM * 2048);     // packed K
  u16* vpbuf  = alloc16((size_t)kM * 2048);     // packed V
  u16* attnb  = alloc16((size_t)kM * 2048);
  (void)ws_size; (void)in_sizes; (void)n_in; (void)out_size;

  auto g1 = [](int n) { return dim3((n + 255) / 256); };

  // precision conversion + weight transposes (L2-resident thereafter)
  f2bf_kernel<<<g1(kM * kHID), 256, 0, stream>>>(x, xb, kM * kHID);
  transpose_f2bf<<<g1(kHID * 512),  256, 0, stream>>>(W_kv_d, wt_kvd, kHID, 512);
  transpose_f2bf<<<g1(kHID * 512),  256, 0, stream>>>(W_q_d,  wt_qd,  kHID, 512);
  transpose_f2bf<<<g1(kLAT * 1024), 256, 0, stream>>>(W_k_u,  wt_ku,  kLAT, 1024);
  transpose_f2bf<<<g1(kLAT * 1024), 256, 0, stream>>>(W_q_u,  wt_qu,  kLAT, 1024);
  transpose_f2bf<<<g1(kLAT * 2048), 256, 0, stream>>>(W_v_u,  wt_vu,  kLAT, 2048);
  transpose_f2bf<<<g1(kHID * 1024), 256, 0, stream>>>(W_rk,   wt_rk,  kHID, 1024);
  transpose_f2bf<<<g1(kLAT * 1024), 256, 0, stream>>>(W_rq,   wt_rq,  kLAT, 1024);
  transpose_f2bf<<<g1(kHID * kHID), 256, 0, stream>>>(W_o,    wt_o,   kHID, kHID);

  // projection GEMMs (bf16 in/out, f32 accumulate)
  gemm_bf16_wmma<true><<<dim3(kM / 128, 512 / 128),  256, 0, stream>>>(xb,  wt_kvd, b_kv_d, kvd, kM, 512,  2048);
  gemm_bf16_wmma<true><<<dim3(kM / 128, 512 / 128),  256, 0, stream>>>(xb,  wt_qd,  b_q_d,  qd,  kM, 512,  2048);
  gemm_bf16_wmma<true><<<dim3(kM / 128, 1024 / 128), 256, 0, stream>>>(kvd, wt_ku,  b_k_u,  k1b, kM, 1024, 512);
  gemm_bf16_wmma<true><<<dim3(kM / 128, 1024 / 128), 256, 0, stream>>>(qd,  wt_qu,  b_q_u,  q1b, kM, 1024, 512);
  gemm_bf16_wmma<true><<<dim3(kM / 128, 2048 / 128), 256, 0, stream>>>(kvd, wt_vu,  b_v_u,  vb,  kM, 2048, 512);
  gemm_bf16_wmma<true><<<dim3(kM / 128, 1024 / 128), 256, 0, stream>>>(xb,  wt_rk,  b_rk,   k2b, kM, 1024, 2048);
  gemm_bf16_wmma<true><<<dim3(kM / 128, 1024 / 128), 256, 0, stream>>>(qd,  wt_rq,  b_rq,   q2b, kM, 1024, 512);

  // RoPE + fragment-native packing of Q/K, then V
  rope_pack<<<g1(kB * kS * kH * 64), 256, 0, stream>>>(q1b, q2b, k1b, k2b, qbuf, kpbuf);
  v_pack<<<g1(kB * kS * kH * kD), 256, 0, stream>>>(vb, vpbuf);

  // causal flash attention (B*H*S/16 wave tiles, 4 waves/block)
  mla_attn<<<dim3(kB * kH * (kS / 16) / 4), 128, 0, stream>>>(qbuf, kpbuf, vpbuf, attnb);

  // output projection, fp32 result
  gemm_bf16_wmma<false><<<dim3(kM / 128, kHID / 128), 256, 0, stream>>>(attnb, wt_o, b_o, d_out, kM, kHID, 2048);
}
